// ViT_29111288332520
// MI455X (gfx1250) — matrix operand order repaired and checked by reading, compiled, not run on hardware
//
#include <hip/hip_runtime.h>

// ---------------- types ----------------
typedef __attribute__((ext_vector_type(16))) __bf16 bf16x16;
typedef __attribute__((ext_vector_type(8)))  float  f32x8;

union Frag { bf16x16 v; uint4 q[2]; };

// ---------------- model dims ----------------
#define BSZ   32
#define SEQ   197
#define NPAT  196
#define EMB   256
#define NH    16
#define HD    16
#define FF    3072
#define NLAY  12
#define MTOK  (BSZ*SEQ)     // 6304
#define MPAT  (BSZ*NPAT)    // 6272

// =====================================================================
// fp32 -> bf16 convert
// =====================================================================
__global__ void f2bf_kernel(const float* __restrict__ in, __bf16* __restrict__ out, long n) {
    long i = (long)blockIdx.x * 256 + threadIdx.x;
    if (i < n) out[i] = (__bf16)in[i];
}

// =====================================================================
// im2col: x (32,1,224,224) -> X (6272, 256) bf16, c = r*16+col
// =====================================================================
__global__ void im2col_kernel(const float* __restrict__ x, __bf16* __restrict__ X) {
    long idx = (long)blockIdx.x * 256 + threadIdx.x;
    if (idx >= (long)MPAT * 256) return;
    int c  = idx & 255;
    int p  = (int)((idx >> 8) % NPAT);
    int b  = (int)(idx / (NPAT * 256));
    int py = p / 14, px = p % 14;
    int r  = c >> 4, col = c & 15;
    X[idx] = (__bf16)x[((long)b * 224 + py * 16 + r) * 224 + px * 16 + col];
}

// =====================================================================
// token assembly: h[b,s,:] = (s==0 ? cls : patch_out[b,s-1]) + pos[s]
// =====================================================================
__global__ void assemble_kernel(const float* __restrict__ patch_out,
                                const float* __restrict__ cls,
                                const float* __restrict__ pos,
                                float* __restrict__ h) {
    long idx = (long)blockIdx.x * 256 + threadIdx.x;
    if (idx >= (long)MTOK * 256) return;
    int  e = idx & 255;
    long t = idx >> 8;
    int  s = (int)(t % SEQ);
    int  b = (int)(t / SEQ);
    float v = (s == 0) ? cls[e] : patch_out[((long)b * NPAT + (s - 1)) * 256 + e];
    h[idx] = v + pos[s * 256 + e];
}

// =====================================================================
// scatter mid -> h rows b=1..30
// =====================================================================
__global__ void midcopy_kernel(const float* __restrict__ mid, float* __restrict__ h) {
    long idx = (long)blockIdx.x * 256 + threadIdx.x;
    if (idx >= (long)(BSZ - 2) * SEQ * 256) return;
    h[idx + (long)SEQ * 256] = mid[idx];
}

// =====================================================================
// generic bf16 WMMA GEMM: out[m,n] = sum_k A[m,k]*W[n,k] + bias[n]
//   optional residual add, exact GELU, f32 and/or bf16 outputs.
// block = 256 threads = 8 waves; wave -> 16(M) x 64(N) tile; K step 32.
// Weight tile (64x32 bf16 = 4KB) is shared by all 8 waves: staged into
// LDS with gfx1250 async-to-LDS loads (ASYNCcnt), double-buffered.
// Per iteration each wave issues exactly ONE async fill, so
// s_wait_asynccnt 0 at loop top == "my part of current tile landed";
// the barrier then makes the whole tile visible, and consumers drain
// their ds reads into the WMMA before reaching the next barrier, so
// refilling the other buffer afterwards is race-free.
// =====================================================================
__global__ __launch_bounds__(256) void gemm_bf16_kernel(
    const __bf16* __restrict__ A, const __bf16* __restrict__ W,
    const float* __restrict__ bias, const float* __restrict__ residual,
    float* __restrict__ outF, __bf16* __restrict__ outB,
    int M, int N, int K, int gelu)
{
    __shared__ __align__(16) __bf16 wtile[2][64 * 32];   // 2 x 4KB weight tiles

    int tid  = threadIdx.x;
    int w    = tid >> 5;
    int lane = tid & 31;
    int half = lane >> 4;
    int l16  = lane & 15;
    int m0 = (blockIdx.y * 8 + w) * 16;
    int n0 = blockIdx.x * 64;
    bool active = (m0 < M);
    int arowi = active ? (m0 + l16) : l16;   // keep inactive waves in-range

    // cooperative async fill: 256 threads cover 64 rows x 4 chunks of 16B
    int frow = tid >> 2;             // 0..63  (weight row within tile)
    int fchk = (tid & 3) * 8;        // 16-byte chunk (element offset)
    const __bf16* gsrc = W + (long)(n0 + frow) * K + fchk;

    f32x8 acc[4] = {};
    const __bf16* arow = A + (long)arowi * K + half * 8;

    int nsteps = K >> 5;
    {   // prologue: fill buffer 0
        unsigned lds = (unsigned)(unsigned long long)&wtile[0][frow * 32 + fchk];
        asm volatile("global_load_async_to_lds_b128 %0, %1, off"
                     :: "v"(lds), "v"((unsigned long long)gsrc) : "memory");
    }

    for (int step = 0; step < nsteps; ++step) {
        asm volatile("s_wait_asynccnt 0x0" ::: "memory");
        __syncthreads();
        if (step + 1 < nsteps) {
            unsigned lds = (unsigned)(unsigned long long)
                &wtile[(step + 1) & 1][frow * 32 + fchk];
            asm volatile("global_load_async_to_lds_b128 %0, %1, off"
                         :: "v"(lds),
                            "v"((unsigned long long)(gsrc + (long)(step + 1) * 32))
                         : "memory");
        }
        int k0 = step << 5;
        Frag a;
        a.q[0] = *(const uint4*)(arow + k0);
        a.q[1] = *(const uint4*)(arow + k0 + 16);
        const __bf16* btile = &wtile[step & 1][0];
#pragma unroll
        for (int j = 0; j < 4; j++) {
            Frag bf;
            const __bf16* bp = btile + (j * 16 + l16) * 32 + half * 8;
            bf.q[0] = *(const uint4*)bp;
            bf.q[1] = *(const uint4*)(bp + 16);
            acc[j] = __builtin_amdgcn_wmma_f32_16x16x32_bf16(
                false, a.v, false, bf.v, (short)0, acc[j], false, false);
        }
    }

    if (!active) return;
#pragma unroll
    for (int j = 0; j < 4; j++) {
        int n = n0 + j * 16 + l16;
        float bsv = bias ? bias[n] : 0.f;
#pragma unroll
        for (int v2 = 0; v2 < 8; v2++) {
            int  m   = m0 + half * 8 + v2;
            long off = (long)m * N + n;
            float val = acc[j][v2] + bsv;
            if (residual) val += residual[off];
            if (gelu)     val = 0.5f * val * (1.f + erff(val * 0.70710678f));
            if (outF) outF[off] = val;
            if (outB) outB[off] = (__bf16)val;
        }
    }
}

// =====================================================================
// LayerNorm (E=256): h f32 -> out bf16
// =====================================================================
__global__ __launch_bounds__(256) void ln_kernel(const float* __restrict__ x,
                                                 const float* __restrict__ g,
                                                 const float* __restrict__ bt,
                                                 __bf16* __restrict__ out, int M) {
    int row = blockIdx.x;
    int e   = threadIdx.x;
    __shared__ float red[256];
    float v = x[(long)row * 256 + e];
    red[e] = v; __syncthreads();
    for (int s = 128; s > 0; s >>= 1) { if (e < s) red[e] += red[e + s]; __syncthreads(); }
    float mean = red[0] * (1.f / 256.f); __syncthreads();
    float d = v - mean;
    red[e] = d * d; __syncthreads();
    for (int s = 128; s > 0; s >>= 1) { if (e < s) red[e] += red[e + s]; __syncthreads(); }
    float var = red[0] * (1.f / 256.f);
    out[(long)row * 256 + e] = (__bf16)(d * rsqrtf(var + 1e-5f) * g[e] + bt[e]);
}

// =====================================================================
// fused multi-head attention per (b, head): qkv bf16 (M,768) -> o bf16 (M,256)
// =====================================================================
__global__ __launch_bounds__(256) void attn_kernel(const __bf16* __restrict__ qkv,
                                                   __bf16* __restrict__ o) {
    int b    = blockIdx.x >> 4;
    int head = blockIdx.x & 15;
    int tid  = threadIdx.x;
    int w    = tid >> 5;
    int lane = tid & 31;
    int half = lane >> 4;
    int l16  = lane & 15;

    __shared__ __align__(16) __bf16 Vt[16][224];     // V transposed (hd, key)
    __shared__ __align__(16) __bf16 P[8][16][224];   // per-wave probabilities

    for (int idx = tid; idx < 16 * 224; idx += 256) {
        int hd = idx / 224, s = idx % 224;
        __bf16 val = (__bf16)0.f;
        if (s < SEQ) val = qkv[((long)(b * SEQ + s)) * 768 + 512 + head * 16 + hd];
        Vt[hd][s] = val;
    }
    __syncthreads();

    for (int qt = w; qt < 13; qt += 8) {
        int s0 = qt * 16;
        // Q fragment (A layout, K padded 16->32 with zeros)
        Frag qa;
        int qs = s0 + l16;
        if (qs < SEQ)
            qa.q[0] = *(const uint4*)(qkv + ((long)(b * SEQ + qs)) * 768 + head * 16 + half * 8);
        else
            qa.q[0] = make_uint4(0, 0, 0, 0);
        qa.q[1] = make_uint4(0, 0, 0, 0);

        f32x8 sc[13] = {};
        for (int kt = 0; kt < 13; kt++) {
            Frag kb;
            int ks = kt * 16 + l16;
            if (ks < SEQ) {
                const __bf16* kp = qkv + ((long)(b * SEQ + ks)) * 768 + 256 + head * 16 + half * 8;
                kb.q[0] = *(const uint4*)kp;
                kb.q[1] = make_uint4(0, 0, 0, 0);
            } else {
                kb.q[0] = make_uint4(0, 0, 0, 0);
                kb.q[1] = make_uint4(0, 0, 0, 0);
            }
            sc[kt] = __builtin_amdgcn_wmma_f32_16x16x32_bf16(
                false, qa.v, false, kb.v, (short)0, sc[kt], false, false);
        }

        // row softmax: row m = half*8+v2, cols striped over 16 lanes x 13 tiles
#pragma unroll
        for (int v2 = 0; v2 < 8; v2++) {
            float mx = -1e30f;
#pragma unroll
            for (int kt = 0; kt < 13; kt++) {
                float xv = sc[kt][v2] * 0.25f;            // 1/sqrt(HD)
                if (kt * 16 + l16 >= SEQ) xv = -1e30f;
                sc[kt][v2] = xv;
                mx = fmaxf(mx, xv);
            }
            for (int off = 1; off < 16; off <<= 1) mx = fmaxf(mx, __shfl_xor(mx, off, 32));
            float sum = 0.f;
#pragma unroll
            for (int kt = 0; kt < 13; kt++) {
                float ev = __expf(sc[kt][v2] - mx);
                sc[kt][v2] = ev;
                sum += ev;
            }
            for (int off = 1; off < 16; off <<= 1) sum += __shfl_xor(sum, off, 32);
            float inv = 1.f / sum;
            int m = half * 8 + v2;
#pragma unroll
            for (int kt = 0; kt < 13; kt++)
                P[w][m][kt * 16 + l16] = (__bf16)(sc[kt][v2] * inv);
            P[w][m][208 + l16] = (__bf16)0.f;            // zero tail pad
        }

        // O = P @ V  (wave-private LDS; compiler inserts ds waits)
        f32x8 oa = {};
        for (int j2 = 0; j2 < 7; j2++) {
            Frag pa, vb;
            pa.q[0] = *(const uint4*)&P[w][l16][j2 * 32 + half * 8];
            pa.q[1] = *(const uint4*)&P[w][l16][j2 * 32 + 16 + half * 8];
            vb.q[0] = *(const uint4*)&Vt[l16][j2 * 32 + half * 8];
            vb.q[1] = *(const uint4*)&Vt[l16][j2 * 32 + 16 + half * 8];
            oa = __builtin_amdgcn_wmma_f32_16x16x32_bf16(
                false, pa.v, false, vb.v, (short)0, oa, false, false);
        }
#pragma unroll
        for (int v2 = 0; v2 < 8; v2++) {
            int s = s0 + half * 8 + v2;
            if (s < SEQ)
                o[((long)(b * SEQ + s)) * 256 + head * 16 + l16] = (__bf16)oa[v2];
        }
    }
}

// =====================================================================
// rotatory attention (rag): one block per i (0..29); tiny FLOPs, plain VALU
// =====================================================================
__device__ __forceinline__ float blk_sum(float v, float* red) {
    int t = threadIdx.x;
    red[t] = v; __syncthreads();
    for (int s = 128; s > 0; s >>= 1) { if (t < s) red[t] += red[t + s]; __syncthreads(); }
    float r = red[0]; __syncthreads();
    return r;
}

__device__ __forceinline__ void blk_softmax(float* p, float* red) {
    int t = threadIdx.x;
    float v = (t < 208) ? p[t] : -1e30f;
    red[t] = v; __syncthreads();
    for (int s = 128; s > 0; s >>= 1) { if (t < s) red[t] = fmaxf(red[t], red[t + s]); __syncthreads(); }
    float mx = red[0]; __syncthreads();
    float e = (t < 208) ? __expf(p[t] - mx) : 0.f;
    red[t] = e; __syncthreads();
    for (int s = 128; s > 0; s >>= 1) { if (t < s) red[t] += red[t + s]; __syncthreads(); }
    float inv = 1.f / red[0]; __syncthreads();
    if (t < 208) p[t] = e * inv;
    __syncthreads();
}

__global__ __launch_bounds__(256) void rag_kernel(
    const float* __restrict__ h,
    const float* __restrict__ Wkl, const float* __restrict__ bkl,
    const float* __restrict__ Wvl, const float* __restrict__ bvl,
    const float* __restrict__ Wkr, const float* __restrict__ bkr,
    const float* __restrict__ Wvr, const float* __restrict__ bvr,
    const float* __restrict__ Wkt, const float* __restrict__ bkt,
    const float* __restrict__ Wvt, const float* __restrict__ bvt,
    float* __restrict__ mid)
{
    int i = blockIdx.x;
    int t = threadIdx.x;
    const float* xl = h + (long)i * SEQ * 256;
    const float* xt = xl + (long)SEQ * 256;
    const float* xr = xt + (long)SEQ * 256;

    __shared__ float rt[256], ml[256], mr[256];
    __shared__ float ul[256], ur[256], utl[256], utr[256];
    __shared__ float pl[208], pr[208], ptl[208], ptr_[208];
    __shared__ float wa0[256], wa1[256], wa2[256], wa3[256];
    __shared__ float fv[208];
    __shared__ float red[256];

    // column means
    float s1 = 0.f, s2 = 0.f, s3 = 0.f;
    for (int s = 0; s < SEQ; s++) {
        s1 += xt[s * 256 + t];
        s2 += xl[s * 256 + t];
        s3 += xr[s * 256 + t];
    }
    rt[t] = s1 * (1.f / SEQ); ml[t] = s2 * (1.f / SEQ); mr[t] = s3 * (1.f / SEQ);
    __syncthreads();

    // u[e] = sum_j vec[j] * W[j*256+e]
    float a_l = 0.f, a_r = 0.f, a_tl = 0.f, a_tr = 0.f;
    for (int j = 0; j < 256; j++) {
        float rtj = rt[j], mlj = ml[j], mrj = mr[j];
        a_l  += rtj * Wkl[j * 256 + t];
        a_r  += rtj * Wkr[j * 256 + t];
        float wkt = Wkt[j * 256 + t];
        a_tl += mlj * wkt;
        a_tr += mrj * wkt;
    }
    ul[t] = a_l; ur[t] = a_r; utl[t] = a_tl; utr[t] = a_tr;
    __syncthreads();

    float c_l  = blk_sum(bkl[t] * rt[t], red);
    float c_r  = blk_sum(bkr[t] * rt[t], red);
    float c_tl = blk_sum(bkt[t] * ml[t], red);
    float c_tr = blk_sum(bkt[t] * mr[t], red);

    const float scale_k = 0.0625f;  // 1/sqrt(E)
    if (t < SEQ) {
        float dl = 0.f, dr = 0.f, dtl = 0.f, dtr = 0.f;
        for (int e = 0; e < 256; e++) {
            dl  += xl[t * 256 + e] * ul[e];
            dr  += xr[t * 256 + e] * ur[e];
            float xte = xt[t * 256 + e];
            dtl += xte * utl[e];
            dtr += xte * utr[e];
        }
        pl[t]   = (dl + c_l) * scale_k;
        pr[t]   = (dr + c_r) * scale_k;
        ptl[t]  = (dtl + c_tl) * scale_k;
        ptr_[t] = (dtr + c_tr) * scale_k;
    } else if (t < 208) {
        pl[t] = -1e30f; pr[t] = -1e30f; ptl[t] = -1e30f; ptr_[t] = -1e30f;
    }
    __syncthreads();

    blk_softmax(pl, red);
    blk_softmax(pr, red);
    blk_softmax(ptl, red);
    blk_softmax(ptr_, red);

    // probability-weighted token averages
    float w1 = 0.f, w2 = 0.f, w3 = 0.f, w4 = 0.f;
    for (int s = 0; s < SEQ; s++) {
        w1 += pl[s]   * xl[s * 256 + t];
        w2 += pr[s]   * xr[s * 256 + t];
        float xte = xt[s * 256 + t];
        w3 += ptl[s]  * xte;
        w4 += ptr_[s] * xte;
    }
    wa0[t] = w1; wa1[t] = w2; wa2[t] = w3; wa3[t] = w4;
    __syncthreads();

    // f = concat(rl(50), rtl(49), rtr(49), rr(49))
    if (t < 50) {
        float a = bvl[t];
        for (int e = 0; e < 256; e++) a += wa0[e] * Wvl[t * 256 + e];
        fv[t] = a;
    } else if (t < 99) {
        int d = t - 50; float a = bvt[d];
        for (int e = 0; e < 256; e++) a += wa2[e] * Wvt[d * 256 + e];
        fv[t] = a;
    } else if (t < 148) {
        int d = t - 99; float a = bvt[d];
        for (int e = 0; e < 256; e++) a += wa3[e] * Wvt[d * 256 + e];
        fv[t] = a;
    } else if (t < SEQ) {
        int d = t - 148; float a = bvr[d];
        for (int e = 0; e < 256; e++) a += wa1[e] * Wvr[d * 256 + e];
        fv[t] = a;
    }
    __syncthreads();

    // mid[i] = outer(f, rt)
    for (long idx = t; idx < (long)SEQ * 256; idx += 256) {
        int e = (int)(idx & 255);
        int s = (int)(idx >> 8);
        mid[((long)i * SEQ + s) * 256 + e] = fv[s] * rt[e];
    }
}

// =====================================================================
// head: LN(cls) @ W_head.T + b_head -> out (32, 8) f32
// =====================================================================
__global__ __launch_bounds__(256) void head_kernel(const float* __restrict__ h,
                                                   const float* __restrict__ g,
                                                   const float* __restrict__ bt,
                                                   const float* __restrict__ Wh,
                                                   const float* __restrict__ bh,
                                                   float* __restrict__ out) {
    int b = blockIdx.x;
    int e = threadIdx.x;
    __shared__ float red[256];
    __shared__ float xn[256];
    float v = h[(long)(b * SEQ) * 256 + e];
    red[e] = v; __syncthreads();
    for (int s = 128; s > 0; s >>= 1) { if (e < s) red[e] += red[e + s]; __syncthreads(); }
    float mean = red[0] * (1.f / 256.f); __syncthreads();
    float d = v - mean;
    red[e] = d * d; __syncthreads();
    for (int s = 128; s > 0; s >>= 1) { if (e < s) red[e] += red[e + s]; __syncthreads(); }
    float var = red[0] * (1.f / 256.f);
    xn[e] = d * rsqrtf(var + 1e-5f) * g[e] + bt[e];
    __syncthreads();
    if (e < 8) {
        float acc = bh[e];
        for (int j = 0; j < 256; j++) acc += xn[j] * Wh[e * 256 + j];
        out[b * 8 + e] = acc;
    }
}

// =====================================================================
// host launcher
// =====================================================================
static inline void* bump(char*& p, size_t bytes) {
    void* r = (void*)p;
    p += (bytes + 255) & ~(size_t)255;
    return r;
}

extern "C" void kernel_launch(void* const* d_in, const int* in_sizes, int n_in,
                              void* d_out, int out_size, void* d_ws, size_t ws_size,
                              hipStream_t stream) {
    (void)in_sizes; (void)n_in; (void)out_size; (void)ws_size;

    const float* x       = (const float*)d_in[0];
    const float* W_patch = (const float*)d_in[1];
    const float* b_patch = (const float*)d_in[2];
    const float* cls_tok = (const float*)d_in[3];
    const float* pos_emb = (const float*)d_in[4];
    const float* Wkl = (const float*)d_in[5];  const float* bkl = (const float*)d_in[6];
    const float* Wvl = (const float*)d_in[7];  const float* bvl = (const float*)d_in[8];
    const float* Wkr = (const float*)d_in[9];  const float* bkr = (const float*)d_in[10];
    const float* Wvr = (const float*)d_in[11]; const float* bvr = (const float*)d_in[12];
    const float* Wkt = (const float*)d_in[13]; const float* bkt = (const float*)d_in[14];
    const float* Wvt = (const float*)d_in[15]; const float* bvt = (const float*)d_in[16];
    const float* ln1_g = (const float*)d_in[17]; const float* ln1_b = (const float*)d_in[18];
    const float* Wqkv  = (const float*)d_in[19]; const float* bqkv  = (const float*)d_in[20];
    const float* Wo    = (const float*)d_in[21]; const float* bo    = (const float*)d_in[22];
    const float* ln2_g = (const float*)d_in[23]; const float* ln2_b = (const float*)d_in[24];
    const float* W1    = (const float*)d_in[25]; const float* b1    = (const float*)d_in[26];
    const float* W2    = (const float*)d_in[27]; const float* b2    = (const float*)d_in[28];
    const float* lnf_g = (const float*)d_in[29]; const float* lnf_b = (const float*)d_in[30];
    const float* W_head = (const float*)d_in[31]; const float* b_head = (const float*)d_in[32];

    // workspace layout
    char* wp = (char*)d_ws;
    const long nWp   = 256L * 256;
    const long nWqkv = (long)NLAY * 768 * 256;
    const long nWo   = (long)NLAY * 256 * 256;
    const long nW1   = (long)NLAY * FF * 256;
    const long nW2   = (long)NLAY * 256 * FF;

    __bf16* Wp_bf   = (__bf16*)bump(wp, nWp * 2);
    __bf16* Wqkv_bf = (__bf16*)bump(wp, nWqkv * 2);
    __bf16* Wo_bf   = (__bf16*)bump(wp, nWo * 2);
    __bf16* W1_bf   = (__bf16*)bump(wp, nW1 * 2);
    __bf16* W2_bf   = (__bf16*)bump(wp, nW2 * 2);
    __bf16* Ximg    = (__bf16*)bump(wp, (long)MPAT * 256 * 2);
    float*  patchf  = (float*) bump(wp, (long)MPAT * 256 * 4);
    float*  h       = (float*) bump(wp, (long)MTOK * 256 * 4);
    float*  mid     = (float*) bump(wp, (long)(BSZ - 2) * SEQ * 256 * 4);
    __bf16* xn      = (__bf16*)bump(wp, (long)MTOK * 256 * 2);
    __bf16* qkvb    = (__bf16*)bump(wp, (long)MTOK * 768 * 2);
    __bf16* ob      = (__bf16*)bump(wp, (long)MTOK * 256 * 2);
    __bf16* ffb     = (__bf16*)bump(wp, (long)MTOK * FF * 2);

    #define GRID1(n) dim3((unsigned)(((n) + 255) / 256))

    // weights -> bf16
    f2bf_kernel<<<GRID1(nWp),   256, 0, stream>>>(W_patch, Wp_bf,   nWp);
    f2bf_kernel<<<GRID1(nWqkv), 256, 0, stream>>>(Wqkv,    Wqkv_bf, nWqkv);
    f2bf_kernel<<<GRID1(nWo),   256, 0, stream>>>(Wo,      Wo_bf,   nWo);
    f2bf_kernel<<<GRID1(nW1),   256, 0, stream>>>(W1,      W1_bf,   nW1);
    f2bf_kernel<<<GRID1(nW2),   256, 0, stream>>>(W2,      W2_bf,   nW2);

    // patch embedding
    im2col_kernel<<<GRID1((long)MPAT * 256), 256, 0, stream>>>(x, Ximg);
    gemm_bf16_kernel<<<dim3(256 / 64, (MPAT + 127) / 128), 256, 0, stream>>>(
        Ximg, Wp_bf, b_patch, nullptr, patchf, nullptr, MPAT, 256, 256, 0);
    assemble_kernel<<<GRID1((long)MTOK * 256), 256, 0, stream>>>(patchf, cls_tok, pos_emb, h);

    // rotatory attention on batch neighbors
    rag_kernel<<<BSZ - 2, 256, 0, stream>>>(h, Wkl, bkl, Wvl, bvl, Wkr, bkr, Wvr, bvr,
                                            Wkt, bkt, Wvt, bvt, mid);
    midcopy_kernel<<<GRID1((long)(BSZ - 2) * SEQ * 256), 256, 0, stream>>>(mid, h);

    // transformer encoder
    for (int l = 0; l < NLAY; l++) {
        ln_kernel<<<MTOK, 256, 0, stream>>>(h, ln1_g + l * 256, ln1_b + l * 256, xn, MTOK);
        gemm_bf16_kernel<<<dim3(768 / 64, (MTOK + 127) / 128), 256, 0, stream>>>(
            xn, Wqkv_bf + (long)l * 768 * 256, bqkv + l * 768,
            nullptr, nullptr, qkvb, MTOK, 768, 256, 0);
        attn_kernel<<<BSZ * NH, 256, 0, stream>>>(qkvb, ob);
        gemm_bf16_kernel<<<dim3(256 / 64, (MTOK + 127) / 128), 256, 0, stream>>>(
            ob, Wo_bf + (long)l * 256 * 256, bo + l * 256,
            h, h, nullptr, MTOK, 256, 256, 0);
        ln_kernel<<<MTOK, 256, 0, stream>>>(h, ln2_g + l * 256, ln2_b + l * 256, xn, MTOK);
        gemm_bf16_kernel<<<dim3(FF / 64, (MTOK + 127) / 128), 256, 0, stream>>>(
            xn, W1_bf + (long)l * FF * 256, b1 + l * FF,
            nullptr, nullptr, ffb, MTOK, FF, 256, 1);
        gemm_bf16_kernel<<<dim3(256 / 64, (MTOK + 127) / 128), 256, 0, stream>>>(
            ffb, W2_bf + (long)l * 256 * FF, b2 + l * 256,
            h, h, nullptr, MTOK, 256, FF, 0);
    }

    // classification head
    head_kernel<<<BSZ, 256, 0, stream>>>(h, lnf_g, lnf_b, W_head, b_head, (float*)d_out);

    #undef GRID1
}
